// LSTMCell_85194971283756
// MI455X (gfx1250) — matrix operand order repaired and checked by reading, not compiled, not run on hardware
//
#include <hip/hip_runtime.h>
#include <hip/hip_bf16.h>
#include <math.h>

typedef __attribute__((ext_vector_type(16))) _Float16 v16h;
typedef __attribute__((ext_vector_type(8)))  float    v8f;

#define B_DIM   4096
#define D_IN    1024
#define D_H     1024
#define K_DIM   (D_IN + D_H)      // 2048
#define TILE_M  128
#define TILE_N  32                // j columns (per gate); WG covers all 4 gates
#define TILE_K  64
#define KTILES  (K_DIM / TILE_K)  // 32
#define LDA     88                // LDS row stride in halves: 176B rows, conflict-free b128 reads

union AFrag { v16h v; uint4 q[2]; };

__global__ __launch_bounds__(256)
void lstm_wmma_kernel(const float* __restrict__ x,
                      const float* __restrict__ hprev,
                      const float* __restrict__ cprev,
                      const float* __restrict__ Wi, const float* __restrict__ bi,
                      const float* __restrict__ Wf, const float* __restrict__ bfv,
                      const float* __restrict__ Wc, const float* __restrict__ bcv,
                      const float* __restrict__ Wo, const float* __restrict__ bov,
                      float* __restrict__ out)
{
    __shared__ __align__(16) _Float16 As[TILE_M * LDA];        // 22.5 KB
    __shared__ __align__(16) _Float16 Ws4[4][TILE_N * LDA];    // 22.5 KB

    const int tid  = threadIdx.x;
    const int lane = tid & 31;
    const int wid  = tid >> 5;        // 0..7
    const int wm   = wid & 3;         // M-wave 0..3 (32 rows each)
    const int wn   = wid >> 2;        // N-wave 0..1 (16 cols each)
    const int g    = lane >> 4;       // lane group (0|1)
    const int l16  = lane & 15;

    const int nblk = blockIdx.x * TILE_N;   // j base, [0,1024)
    const int mblk = blockIdx.y * TILE_M;   // m base, [0,4096)

    const float* Wq[4] = { Wi, Wf, Wc, Wo };

    // per-thread staging coords: 16 threads cover one 64-float row segment
    const int tr   = tid >> 4;        // 0..15
    const int kcol = (tid & 15) * 4;  // 0..60

    // ---- staging helpers (global f32 -> regs -> LDS f16) ----
    auto loadA = [&](int kt, float4* ra) {
        const int kb = kt * TILE_K;
        const float* src = (kb < D_IN)
            ? (x     + (size_t)mblk * D_IN + kb)
            : (hprev + (size_t)mblk * D_H  + (kb - D_IN));
#pragma unroll
        for (int i = 0; i < 8; ++i)   // 128 rows / 16 row-groups
            ra[i] = *(const float4*)(src + (size_t)(tr + 16 * i) * 1024 + kcol);
    };
    auto loadW = [&](int kt, float4* rw) {
        const int kb = kt * TILE_K;
#pragma unroll
        for (int q = 0; q < 4; ++q) {
            const float* src = Wq[q] + (size_t)nblk * K_DIM + kb;
#pragma unroll
            for (int ii = 0; ii < 2; ++ii)
                rw[q * 2 + ii] = *(const float4*)(src + (size_t)(tr + 16 * ii) * K_DIM + kcol);
        }
    };
    auto stA = [&](const float4* ra) {
#pragma unroll
        for (int i = 0; i < 8; ++i) {
            _Float16* p = &As[(tr + 16 * i) * LDA + kcol];
            p[0] = (_Float16)ra[i].x; p[1] = (_Float16)ra[i].y;
            p[2] = (_Float16)ra[i].z; p[3] = (_Float16)ra[i].w;
        }
    };
    auto stW = [&](const float4* rw) {
#pragma unroll
        for (int q = 0; q < 4; ++q)
#pragma unroll
            for (int ii = 0; ii < 2; ++ii) {
                const float4 v = rw[q * 2 + ii];
                _Float16* p = &Ws4[q][(tr + 16 * ii) * LDA + kcol];
                p[0] = (_Float16)v.x; p[1] = (_Float16)v.y;
                p[2] = (_Float16)v.z; p[3] = (_Float16)v.w;
            }
    };

    // ---- prologue: stage tile 0 ----
    {
        float4 ra[8], rw[8];
        loadA(0, ra); loadW(0, rw);
        stA(ra);      stW(rw);
    }

    // acc[gate][m-frag] : 16x16 f32 each; 64 VGPRs total
    v8f acc[4][2] = {};

    for (int kt = 0; kt < KTILES; ++kt) {
        __syncthreads();                       // LDS tile kt visible

        float4 na[8], nw[8];
        if (kt + 1 < KTILES) { loadA(kt + 1, na); loadW(kt + 1, nw); }

        // ---- compute tile kt: two k32 sub-steps; 2 M-frags x 4 gates ----
#pragma unroll
        for (int ks = 0; ks < TILE_K; ks += 32) {
            AFrag a[2];
#pragma unroll
            for (int f = 0; f < 2; ++f) {
                const _Float16* ap = &As[(32 * wm + 16 * f + l16) * LDA + ks + g * 8];
                a[f].q[0] = *(const uint4*)(ap);        // K = ks +      g*8 .. +7
                a[f].q[1] = *(const uint4*)(ap + 16);   // K = ks + 16 + g*8 .. +7
            }
#pragma unroll
            for (int q = 0; q < 4; ++q) {
                AFrag b;
                const _Float16* bp = &Ws4[q][(16 * wn + l16) * LDA + ks + g * 8];
                b.q[0] = *(const uint4*)(bp);           // K = ks +      g*8 .. +7
                b.q[1] = *(const uint4*)(bp + 16);      // K = ks + 16 + g*8 .. +7
#pragma unroll
                for (int f = 0; f < 2; ++f)
                    acc[q][f] = __builtin_amdgcn_wmma_f32_16x16x32_f16(
                        false, a[f].v, false, b.v, (short)0, acc[q][f], false, false);
            }
        }

        __syncthreads();                       // everyone done reading tile kt
        if (kt + 1 < KTILES) { stA(na); stW(nw); }
    }

    // ---- fused LSTM epilogue ----
    const int j = nblk + 16 * wn + l16;        // hidden index [0,1024)
    const float Bi = bi[j], Bf = bfv[j], Bc = bcv[j], Bo = bov[j];

#pragma unroll
    for (int f = 0; f < 2; ++f) {
        const int m0 = mblk + 32 * wm + 16 * f + 8 * g;
#pragma unroll
        for (int r = 0; r < 8; ++r) {
            const int m = m0 + r;
            const float zi = acc[0][f][r] + Bi;
            const float zf = acc[1][f][r] + Bf;
            const float zc = acc[2][f][r] + Bc;
            const float zo = acc[3][f][r] + Bo;
            const float ig = 1.0f / (1.0f + __expf(-zi));
            const float fg = 1.0f / (1.0f + __expf(-zf));
            const float cg = tanhf(zc);
            const float og = 1.0f / (1.0f + __expf(-zo));
            const float cp = cprev[(size_t)m * D_H + j];
            const float cn = fg * cp + ig * cg;
            const float hn = og * tanhf(cn);
            out[(size_t)m * D_H + j] = hn;                           // h_next
            out[(size_t)B_DIM * D_H + (size_t)m * D_H + j] = cn;     // c_next
        }
    }
}

extern "C" void kernel_launch(void* const* d_in, const int* in_sizes, int n_in,
                              void* d_out, int out_size, void* d_ws, size_t ws_size,
                              hipStream_t stream) {
    (void)in_sizes; (void)n_in; (void)out_size; (void)d_ws; (void)ws_size;
    const float* x     = (const float*)d_in[0];
    const float* hprev = (const float*)d_in[1];
    const float* cprev = (const float*)d_in[2];
    const float* Wi    = (const float*)d_in[3];
    const float* bi    = (const float*)d_in[4];
    const float* Wf    = (const float*)d_in[5];
    const float* bf_   = (const float*)d_in[6];
    const float* Wc    = (const float*)d_in[7];
    const float* bc_   = (const float*)d_in[8];
    const float* Wo    = (const float*)d_in[9];
    const float* bo_   = (const float*)d_in[10];
    float* out = (float*)d_out;

    dim3 grid(D_H / TILE_N, B_DIM / TILE_M);   // (32, 32)
    dim3 block(256);
    lstm_wmma_kernel<<<grid, block, 0, stream>>>(x, hprev, cprev,
                                                 Wi, bi, Wf, bf_, Wc, bc_, Wo, bo_,
                                                 out);
}
